// Attention_36696200577520
// MI455X (gfx1250) — matrix operand order repaired and checked by reading, not compiled, not run on hardware
//
#include <hip/hip_runtime.h>
#include <hip/hip_bf16.h>

#define DMODEL 1024
#define DK     64
#define DHEAD  16
#define DEPTH  1024
#define NBATCH 2
#define SEQ    2048
#define LDSP   40   // padded LDS row stride (elements): conflict-free fragment reads

#if defined(__has_builtin)
#if __has_builtin(__builtin_amdgcn_global_load_async_to_lds_b128) && \
    __has_builtin(__builtin_amdgcn_s_wait_asynccnt)
#define HAVE_ASYNC_LDS 1
#endif
#endif

typedef __attribute__((ext_vector_type(2)))  __bf16 v2bf;
typedef __attribute__((ext_vector_type(8)))  __bf16 v8bf;
typedef __attribute__((ext_vector_type(16))) __bf16 v16bf;
typedef __attribute__((ext_vector_type(4)))  float  v4f;
typedef __attribute__((ext_vector_type(8)))  float  v8f;
typedef __attribute__((ext_vector_type(4)))  int    v4i;

typedef __attribute__((address_space(1))) v4i GV4;   // global 128-bit chunk
typedef __attribute__((address_space(3))) v4i LV4;   // LDS 128-bit chunk

__device__ __forceinline__ __bf16 f2bf(float f) {
  unsigned u = __builtin_bit_cast(unsigned, f);
  u += 0x7fffu + ((u >> 16) & 1u);               // round-to-nearest-even
  unsigned short h = (unsigned short)(u >> 16);
  return __builtin_bit_cast(__bf16, h);
}

// two f32 -> packed 2x bf16 (RNE) in one 32-bit value
__device__ __forceinline__ unsigned pack2u(float a, float b) {
  unsigned ua = __builtin_bit_cast(unsigned, a);
  unsigned ub = __builtin_bit_cast(unsigned, b);
  ua += 0x7fffu + ((ua >> 16) & 1u);
  ub += 0x7fffu + ((ub >> 16) & 1u);
  return (ua >> 16) | (ub & 0xffff0000u);
}

__device__ __forceinline__ v16bf cat16(v8bf lo, v8bf hi) {
  return __builtin_shufflevector(lo, hi, 0,1,2,3,4,5,6,7,8,9,10,11,12,13,14,15);
}

// A-fragment (16x32 bf16) from row-major source, row stride ld (elements).
// Lane L: row = L%16; elements j: K = (L/16)*16 + j  -> two 16B runs.
__device__ __forceinline__ v16bf load_a_frag(const __bf16* base, int ld) {
  const int lane = threadIdx.x & 31;
  const __bf16* p = base + (lane & 15) * ld + (lane >> 4) * 16;
  return cat16(*(const v8bf*)p, *(const v8bf*)(p + 8));
}

// B-fragment: per-lane 32 contiguous bytes (16B-aligned is sufficient)
__device__ __forceinline__ v16bf load_b_frag(const __bf16* p) {
  return cat16(*(const v8bf*)p, *(const v8bf*)(p + 8));
}

__device__ __forceinline__ v8f wmma_bf16(v16bf a, v16bf b, v8f c) {
  return __builtin_amdgcn_wmma_f32_16x16x32_bf16(false, a, false, b, (short)0, c,
                                                 false, false);
}

// async copy of one 16B chunk global -> LDS (falls back to VGPR round-trip)
__device__ __forceinline__ void copy16_g2l(void* lds, const void* g) {
#ifdef HAVE_ASYNC_LDS
  __builtin_amdgcn_global_load_async_to_lds_b128((GV4*)g, (LV4*)lds, 0, 0);
#else
  *(v8bf*)lds = *(const v8bf*)g;
#endif
}

// ---------------------------------------------------------------------------
// Pre-pass 1: elementwise f32 -> bf16 for x (blockIdx.y=0) and y (=1).
// ---------------------------------------------------------------------------
__global__ __launch_bounds__(256)
void conv_xy_kernel(const float* __restrict__ x, const float* __restrict__ y,
                    __bf16* __restrict__ xbf, __bf16* __restrict__ ybf) {
  const float* src = blockIdx.y ? y : x;
  __bf16* dst = blockIdx.y ? ybf : xbf;
  size_t i8 = ((size_t)blockIdx.x * 256 + threadIdx.x) * 8;
  v4f a = *(const v4f*)&src[i8];
  v4f b = *(const v4f*)&src[i8 + 4];
  union { unsigned u[4]; v8bf v; } t;
  t.u[0] = pack2u(a[0], a[1]);
  t.u[1] = pack2u(a[2], a[3]);
  t.u[2] = pack2u(b[0], b[1]);
  t.u[3] = pack2u(b[2], b[3]);
  *(v8bf*)&dst[i8] = t.v;
}

// ---------------------------------------------------------------------------
// Pre-pass 2: convert + transpose weights: Wt[n][k] (bf16) = W[k][n] (f32).
// blockIdx.z selects which of the 4 weight matrices. 64x64 tiles.
// ---------------------------------------------------------------------------
__global__ __launch_bounds__(256)
void conv_wt_kernel(const float* __restrict__ Wq, const float* __restrict__ Wk,
                    const float* __restrict__ Wv, const float* __restrict__ Wo,
                    __bf16* __restrict__ Wt) {
  const int z = blockIdx.z;
  const float* W = (z == 0) ? Wq : (z == 1) ? Wk : (z == 2) ? Wv : Wo;
  __bf16* T = Wt + (size_t)z * DMODEL * DEPTH;
  const int k0 = blockIdx.x * 64;
  const int n0 = blockIdx.y * 64;
  const int tid = threadIdx.x;
  __shared__ __bf16 Tl[64][65];
#pragma unroll
  for (int i = 0; i < 16; ++i) {                 // coalesced load, convert
    int idx = tid + i * 256;
    int r = idx >> 6, c = idx & 63;
    Tl[r][c] = f2bf(W[(size_t)(k0 + r) * DMODEL + n0 + c]);
  }
  __syncthreads();
#pragma unroll
  for (int i = 0; i < 8; ++i) {                  // coalesced packed store
    int idx = tid + i * 256;                     // 2048 pairs
    int rr = idx >> 5, cc2 = (idx & 31) * 2;
    unsigned pr =
        (unsigned)__builtin_bit_cast(unsigned short, Tl[cc2][rr]) |
        ((unsigned)__builtin_bit_cast(unsigned short, Tl[cc2 + 1][rr]) << 16);
    *(v2bf*)&T[(size_t)(n0 + rr) * DMODEL + k0 + cc2] =
        __builtin_bit_cast(v2bf, pr);
  }
}

// ---------------------------------------------------------------------------
// GEMM B-tile staging: 128x32 bf16 tile of Wt[n][k] -> LDS via async copies.
// ---------------------------------------------------------------------------
__device__ __forceinline__ void stage_b(__bf16 (*Blds)[LDSP], const __bf16* Wt,
                                        int n0, int k0, int tid) {
#pragma unroll
  for (int i = 0; i < 2; ++i) {
    int q = tid + i * 256;                       // 512 chunks of 16B
    int n = q >> 2, j = q & 3;
    copy16_g2l((char*)&Blds[0][0] + n * (LDSP * 2) + j * 16,
               (const char*)Wt + ((size_t)(n0 + n) * DMODEL + k0) * 2 + j * 16);
  }
}

// ---------------------------------------------------------------------------
// Kernel 1: QKV projections. blockIdx.z: 0=Q(xbf) 1=K(ybf) 2=V(ybf).
// 128x128 tiles, 256 threads (8 wave32), A fragments straight from global,
// B tiles async-staged in LDS with double buffering.
// Q,K written bf16 [bh][l][64]; V written transposed bf16 [bh][64][l].
// ---------------------------------------------------------------------------
__global__ __launch_bounds__(256)
void qkv_proj_kernel(const __bf16* __restrict__ xbf, const __bf16* __restrict__ ybf,
                     const __bf16* __restrict__ Wt,  // 3 transposed weights
                     const float* __restrict__ bq, const float* __restrict__ bk,
                     const float* __restrict__ bv,
                     __bf16* __restrict__ Qb, __bf16* __restrict__ Kb,
                     __bf16* __restrict__ Vt) {
  const int z = blockIdx.z;
  const __bf16* A   = (z == 0) ? xbf : ybf;
  const __bf16* Wz  = Wt + (size_t)z * DMODEL * DEPTH;
  const float* bias = (z == 0) ? bq : (z == 1) ? bk : bv;

  const int n0 = blockIdx.x * 128;
  const int m0 = blockIdx.y * 128;
  const int tid  = threadIdx.x;
  const int w    = tid >> 5;
  const int lane = tid & 31;
  const int half = lane >> 4;
  const int lx   = lane & 15;
  const int rowg = (w & 3) * 32;
  const int colg = (w >> 2) * 64;

  __shared__ alignas(32) __bf16 Blds[2][128][LDSP];

  v8f acc0[4], acc1[4];
#pragma unroll
  for (int c = 0; c < 4; ++c)
#pragma unroll
    for (int v = 0; v < 8; ++v) { acc0[c][v] = 0.0f; acc1[c][v] = 0.0f; }

  const __bf16* a0p = A + (size_t)(m0 + rowg) * DMODEL;
  const __bf16* a1p = A + (size_t)(m0 + rowg + 16) * DMODEL;

  stage_b(Blds[0], Wz, n0, 0, tid);              // preload tile 0
  for (int kb = 0; kb < DMODEL / 32; ++kb) {
    const int k0 = kb * 32;
    if (kb + 1 < DMODEL / 32) {
      stage_b(Blds[(kb + 1) & 1], Wz, n0, k0 + 32, tid);
#ifdef HAVE_ASYNC_LDS
      __builtin_amdgcn_s_wait_asynccnt(2);       // tile kb landed (next in flight)
#endif
    } else {
#ifdef HAVE_ASYNC_LDS
      __builtin_amdgcn_s_wait_asynccnt(0);
#endif
    }
    __syncthreads();
    v16bf a0 = load_a_frag(a0p + k0, DMODEL);
    v16bf a1 = load_a_frag(a1p + k0, DMODEL);
    const __bf16(*Bt)[LDSP] = Blds[kb & 1];
#pragma unroll
    for (int c = 0; c < 4; ++c) {
      v16bf bfrag = load_b_frag(&Bt[colg + c * 16 + lx][half * 16]);
      acc0[c] = wmma_bf16(a0, bfrag, acc0[c]);
      acc1[c] = wmma_bf16(a1, bfrag, acc1[c]);
    }
    __syncthreads();
  }

#pragma unroll
  for (int rh = 0; rh < 2; ++rh) {
#pragma unroll
    for (int c = 0; c < 4; ++c) {
      v8f accv;
#pragma unroll
      for (int v = 0; v < 8; ++v) accv[v] = rh ? acc1[c][v] : acc0[c][v];
      const int ncol = n0 + colg + c * 16 + lx;
      const float bb = bias[ncol];
      const int h = ncol >> 6;
      const int d = ncol & 63;
      if (z == 2) {                              // V transposed: 8 contiguous l
        int m = m0 + rowg + rh * 16 + half * 8;
        int batch = m >> 11, l = m & 2047;
        v8bf t;
#pragma unroll
        for (int v = 0; v < 8; ++v) t[v] = f2bf(accv[v] + bb);
        *(v8bf*)&Vt[((size_t)(batch * 16 + h) * 64 + d) * SEQ + l] = t;
      } else {
        __bf16* dst = (z == 0) ? Qb : Kb;
#pragma unroll
        for (int v = 0; v < 8; ++v) {
          int m = m0 + rowg + rh * 16 + half * 8 + v;
          int batch = m >> 11, l = m & 2047;
          dst[((size_t)(batch * 16 + h) * SEQ + l) * 64 + d] = f2bf(accv[v] + bb);
        }
      }
    }
  }
}

// ---------------------------------------------------------------------------
// Kernel 2: causal flash attention. Block = 4 waves, 64 query rows per block,
// 16 per wave. Key loop in blocks of 32 keys, online softmax.
// ---------------------------------------------------------------------------
__global__ __launch_bounds__(128)
void attn_kernel(const __bf16* __restrict__ Qb, const __bf16* __restrict__ Kb,
                 const __bf16* __restrict__ Vt, __bf16* __restrict__ ctx) {
  const int bh    = blockIdx.y;                  // batch*16 + head
  const int batch = bh >> 4, h = bh & 15;
  const int tid   = threadIdx.x;
  const int w     = tid >> 5;
  const int lane  = tid & 31;
  const int half  = lane >> 4;
  const int lx    = lane & 15;
  const int q0    = blockIdx.x * 64 + w * 16;    // first query row for this wave

  const __bf16* Qh = Qb + (size_t)bh * SEQ * 64;
  const __bf16* Kh = Kb + (size_t)bh * SEQ * 64;
  const __bf16* Vh = Vt + (size_t)bh * 64 * SEQ;

  __shared__ alignas(32) __bf16 Plds[4][16][LDSP]; // per-wave P staging

  v16bf qa0 = load_a_frag(Qh + (size_t)q0 * 64 + 0, 64);
  v16bf qa1 = load_a_frag(Qh + (size_t)q0 * 64 + 32, 64);

  v8f O[4];
  float mrow[8], lrow[8];
#pragma unroll
  for (int c = 0; c < 4; ++c)
#pragma unroll
    for (int v = 0; v < 8; ++v) O[c][v] = 0.0f;
#pragma unroll
  for (int v = 0; v < 8; ++v) { mrow[v] = -1e30f; lrow[v] = 0.0f; }

  const int kEnd = q0 + 16;                      // causal: keys < kEnd needed
  const int nkb  = (kEnd + 31) >> 5;
  for (int kb = 0; kb < nkb; ++kb) {
    const int kbase = kb * 32;

    if (kb + 1 < nkb) {                          // pull next key block toward L0
      __builtin_prefetch(Kh + (size_t)(kbase + 32 + lx) * 64, 0, 3);
      __builtin_prefetch(Vh + (size_t)(lx * 4) * SEQ + kbase + 32, 0, 3);
    }

    // ---- scores: S[s] = Q(16x64) x K^T(64x16), keys kbase+16s .. +16
    v8f S[2];
#pragma unroll
    for (int s = 0; s < 2; ++s) {
      const __bf16* kp = Kh + (size_t)(kbase + s * 16 + lx) * 64 + half * 16;
      v16bf b0 = load_b_frag(kp);                // kdim 0..31 slice
      v16bf b1 = load_b_frag(kp + 32);           // kdim 32..63 slice
      v8f sa;
#pragma unroll
      for (int v = 0; v < 8; ++v) sa[v] = 0.0f;
      sa = wmma_bf16(qa0, b0, sa);
      sa = wmma_bf16(qa1, b1, sa);
      S[s] = sa;
    }

    // ---- scale + causal mask
#pragma unroll
    for (int s = 0; s < 2; ++s)
#pragma unroll
      for (int v = 0; v < 8; ++v) {
        int qrow = q0 + v + 8 * half;
        int kcol = kbase + s * 16 + lx;
        float val = S[s][v] * 0.125f;            // 1/sqrt(64)
        S[s][v] = (kcol <= qrow) ? val : -1e9f;
      }

    // ---- online softmax (rows live across 16-lane halves)
    float mnew[8], alpha[8];
#pragma unroll
    for (int v = 0; v < 8; ++v) {
      float t = fmaxf(S[0][v], S[1][v]);
      t = fmaxf(t, __shfl_xor(t, 1, 32));
      t = fmaxf(t, __shfl_xor(t, 2, 32));
      t = fmaxf(t, __shfl_xor(t, 4, 32));
      t = fmaxf(t, __shfl_xor(t, 8, 32));
      mnew[v]  = fmaxf(mrow[v], t);
      alpha[v] = __expf(mrow[v] - mnew[v]);
      mrow[v]  = mnew[v];
    }
#pragma unroll
    for (int v = 0; v < 8; ++v) {
      S[0][v] = __expf(S[0][v] - mnew[v]);
      S[1][v] = __expf(S[1][v] - mnew[v]);
      float t = S[0][v] + S[1][v];
      t += __shfl_xor(t, 1, 32);
      t += __shfl_xor(t, 2, 32);
      t += __shfl_xor(t, 4, 32);
      t += __shfl_xor(t, 8, 32);
      lrow[v] = lrow[v] * alpha[v] + t;
    }
#pragma unroll
    for (int c = 0; c < 4; ++c)
#pragma unroll
      for (int v = 0; v < 8; ++v) O[c][v] *= alpha[v];

    // ---- transpose P (C-layout) -> A-layout via per-wave LDS
#pragma unroll
    for (int v = 0; v < 8; ++v) {
      int r = v + 8 * half;
      Plds[w][r][lx]      = f2bf(S[0][v]);
      Plds[w][r][16 + lx] = f2bf(S[1][v]);
    }
    asm volatile("s_wait_dscnt 0" ::: "memory"); // same-wave LDS RAW fence
    v16bf pfrag = load_a_frag(&Plds[w][0][0], LDSP);

    // ---- O += P(16x32) x V(32x64), Vt gives contiguous B-fragments
#pragma unroll
    for (int c = 0; c < 4; ++c) {
      v16bf vb = load_b_frag(Vh + (size_t)(c * 16 + lx) * SEQ + kbase + half * 16);
      O[c] = wmma_bf16(pfrag, vb, O[c]);
    }
  }

  // ---- normalize and write context [m][h*64+d] as bf16
  float linv[8];
#pragma unroll
  for (int v = 0; v < 8; ++v) linv[v] = 1.0f / lrow[v];
#pragma unroll
  for (int c = 0; c < 4; ++c) {
    int d = h * 64 + c * 16 + lx;
#pragma unroll
    for (int v = 0; v < 8; ++v) {
      int qrow = q0 + v + 8 * half;
      size_t m = (size_t)batch * SEQ + qrow;
      ctx[m * DEPTH + d] = f2bf(O[c][v] * linv[v]);
    }
  }
}

// ---------------------------------------------------------------------------
// Kernel 3: output projection out = ctx(bf16) @ Wo + bo (f32 out), using the
// pre-transposed Wot with async-staged B tiles.
// ---------------------------------------------------------------------------
__global__ __launch_bounds__(256)
void out_proj_kernel(const __bf16* __restrict__ A, const __bf16* __restrict__ Wot,
                     const float* __restrict__ bo, float* __restrict__ out) {
  const int n0 = blockIdx.x * 128;
  const int m0 = blockIdx.y * 128;
  const int tid  = threadIdx.x;
  const int w    = tid >> 5;
  const int lane = tid & 31;
  const int half = lane >> 4;
  const int lx   = lane & 15;
  const int rowg = (w & 3) * 32;
  const int colg = (w >> 2) * 64;

  __shared__ alignas(32) __bf16 Blds[2][128][LDSP];

  v8f acc0[4], acc1[4];
#pragma unroll
  for (int c = 0; c < 4; ++c)
#pragma unroll
    for (int v = 0; v < 8; ++v) { acc0[c][v] = 0.0f; acc1[c][v] = 0.0f; }

  const __bf16* a0p = A + (size_t)(m0 + rowg) * DEPTH;
  const __bf16* a1p = A + (size_t)(m0 + rowg + 16) * DEPTH;

  stage_b(Blds[0], Wot, n0, 0, tid);
  for (int kb = 0; kb < DEPTH / 32; ++kb) {
    const int k0 = kb * 32;
    if (kb + 1 < DEPTH / 32) {
      stage_b(Blds[(kb + 1) & 1], Wot, n0, k0 + 32, tid);
#ifdef HAVE_ASYNC_LDS
      __builtin_amdgcn_s_wait_asynccnt(2);
#endif
    } else {
#ifdef HAVE_ASYNC_LDS
      __builtin_amdgcn_s_wait_asynccnt(0);
#endif
    }
    __syncthreads();
    v16bf a0 = load_a_frag(a0p + k0, DEPTH);
    v16bf a1 = load_a_frag(a1p + k0, DEPTH);
    const __bf16(*Bt)[LDSP] = Blds[kb & 1];
#pragma unroll
    for (int c = 0; c < 4; ++c) {
      v16bf bfrag = load_b_frag(&Bt[colg + c * 16 + lx][half * 16]);
      acc0[c] = wmma_bf16(a0, bfrag, acc0[c]);
      acc1[c] = wmma_bf16(a1, bfrag, acc1[c]);
    }
    __syncthreads();
  }

#pragma unroll
  for (int rh = 0; rh < 2; ++rh) {
#pragma unroll
    for (int c = 0; c < 4; ++c) {
      int ncol = n0 + colg + c * 16 + lx;
      float bb = bo[ncol];
#pragma unroll
      for (int v = 0; v < 8; ++v) {
        float val = (rh ? acc1[c][v] : acc0[c][v]) + bb;
        int m = m0 + rowg + rh * 16 + v + 8 * half;
        out[(size_t)m * DMODEL + ncol] = val;
      }
    }
  }
}

// ---------------------------------------------------------------------------
extern "C" void kernel_launch(void* const* d_in, const int* in_sizes, int n_in,
                              void* d_out, int out_size, void* d_ws, size_t ws_size,
                              hipStream_t stream) {
  (void)in_sizes; (void)n_in; (void)out_size; (void)ws_size;
  const float* x  = (const float*)d_in[0];
  const float* y  = (const float*)d_in[1];
  // d_in[2] = mask (causal, handled analytically)
  const float* Wq = (const float*)d_in[3];
  const float* bq = (const float*)d_in[4];
  const float* Wk = (const float*)d_in[5];
  const float* bk = (const float*)d_in[6];
  const float* Wv = (const float*)d_in[7];
  const float* bv = (const float*)d_in[8];
  const float* Wo = (const float*)d_in[9];
  const float* bo = (const float*)d_in[10];
  float* out = (float*)d_out;

  const size_t M1 = (size_t)1024 * 1024;
  __bf16* ybf = (__bf16*)d_ws;         // 4M elems
  __bf16* Wt  = ybf + 4 * M1;          // 4 x 1M (Wqt,Wkt,Wvt,Wot)
  __bf16* Qb  = Wt + 4 * M1;           // 4M
  __bf16* Kb  = Qb + 4 * M1;           // 4M
  __bf16* Vt  = Kb + 4 * M1;           // 4M
  __bf16* xbf = Vt + 4 * M1;           // 4M; reused as ctx after qkv proj
  __bf16* ctx = xbf;

  dim3 gc((NBATCH * SEQ * DMODEL) / (256 * 8), 2);        // 2048 x 2
  conv_xy_kernel<<<gc, 256, 0, stream>>>(x, y, xbf, ybf);

  dim3 gw(DMODEL / 64, DEPTH / 64, 4);                    // 16 x 16 x 4
  conv_wt_kernel<<<gw, 256, 0, stream>>>(Wq, Wk, Wv, Wo, Wt);

  dim3 g1(DEPTH / 128, (NBATCH * SEQ) / 128, 3);          // 8 x 32 x 3
  qkv_proj_kernel<<<g1, 256, 0, stream>>>(xbf, ybf, Wt, bq, bk, bv, Qb, Kb, Vt);

  dim3 g2(SEQ / 64, NBATCH * DHEAD);                      // 32 x 32
  attn_kernel<<<g2, 128, 0, stream>>>(Qb, Kb, Vt, ctx);

  dim3 g3(DMODEL / 128, (NBATCH * SEQ) / 128);            // 8 x 32
  out_proj_kernel<<<g3, 256, 0, stream>>>(ctx, Wt + 3 * M1, bo, out);
}
